// GCAModel_22514218566027
// MI455X (gfx1250) — matrix operand order repaired and checked by reading, not compiled, not run on hardware
//
#include <hip/hip_runtime.h>
#include <stdint.h>

// Problem constants (from the reference)
#define Tn 100
#define Jn 25
#define Bn 128
#define In 3
#define Hn 256
#define Cn 60
#define GH (5 * Hn)        // 1280 gate rows
#define NSLICE 8           // hidden-dim slices -> 25*8 = 200 blocks/diagonal
#define HS (Hn / NSLICE)   // 32 hidden per slice
#define NCOL (5 * HS)      // 160 local gate columns per block
#define NT (NCOL / 16)     // 10 WMMA column tiles

#define USE_ASYNC_LDS 1

typedef unsigned short u16;
typedef __attribute__((ext_vector_type(16))) __bf16 v16bf;
typedef __attribute__((ext_vector_type(8)))  float  v8f;
typedef __attribute__((ext_vector_type(4)))  unsigned int u32x4;
typedef __attribute__((ext_vector_type(8)))  int i32x8;
typedef __attribute__((ext_vector_type(4)))  int i32x4;

__device__ __forceinline__ u16 f2bf(float f) {
  uint32_t x = __float_as_uint(f);
  uint32_t r = x + 0x7fffu + ((x >> 16) & 1u);   // RNE
  return (u16)(r >> 16);
}

__device__ __forceinline__ float sigmoidf_(float x) {
  return 1.0f / (1.0f + __expf(-x));
}

// ---- async global -> LDS copy (CDNA5 ASYNCcnt path), 16B per lane ----
__device__ __forceinline__ void cp_async16(u16* ldst, const u16* gsrc) {
#if USE_ASYNC_LDS
  uint32_t l = (uint32_t)(uintptr_t)ldst;        // low 32 bits == LDS byte address
  uint64_t g = (uint64_t)(uintptr_t)gsrc;
  asm volatile("global_load_async_to_lds_b128 %0, %1, off"
               :: "v"(l), "v"(g) : "memory");
#else
  *(uint4*)ldst = *(const uint4*)gsrc;
#endif
}
__device__ __forceinline__ void async_wait_all() {
#if USE_ASYNC_LDS
  asm volatile("s_wait_asynccnt 0" ::: "memory");
#endif
}

// ---- TDM: one 3-D descriptor moves a whole W slice (5 gates x 32 rows x 256 cols,
// bf16) from global into LDS, contiguously in our [r=g*32+hh][k] layout ----
__device__ __forceinline__ void tdm_load_w_slice(u16* lds_dst, const u16* gbase) {
#if __has_builtin(__builtin_amdgcn_tensor_load_to_lds)
  const uint64_t ga = (uint64_t)(uintptr_t)gbase;           // byte address
  const uint32_t la = (uint32_t)(uintptr_t)lds_dst;         // LDS byte address
  // D# group0: count=1 | lds_addr | global_addr(57b) | type=2
  u32x4 g0 = { 1u, la, (uint32_t)ga, (uint32_t)(ga >> 32) | (2u << 30) };
  // D# group1: data_size=2B; tensor_dim0=256; tensor_dim1=32; tile=(256,32,5);
  //            dim0_stride=256; dim1_stride=65536 (gate plane = 256 rows)
  i32x8 g1 = { (int)(1u << 16),            // workgroup_mask=0, data_size=1 (2B)
               (int)(256u << 16),          // tensor_dim0[15:0] -> bits 63:48
               (int)(32u << 16),           // dim0[31:16]=0 | tensor_dim1[15:0]
               (int)(256u << 16),          // dim1[31:16]=0 | tile_dim0=256
               (int)(32u | (5u << 16)),    // tile_dim1=32 | tile_dim2=5
               (int)256,                   // tensor_dim0_stride[31:0]
               (int)0,                     // dim0_stride[47:32] | dim1_stride[15:0]
               (int)1 };                   // tensor_dim1_stride[47:16] (65536>>16)
  // D# group2: tensor_dim2=5; rest unused
  i32x4 g2 = { 5, 0, 0, 0 };
  i32x4 g3 = { 0, 0, 0, 0 };
  i32x8 g4 = { 0, 0, 0, 0, 0, 0, 0, 0 };   // extra word group (6-arg toolchain), unused
  __builtin_amdgcn_tensor_load_to_lds(g0, g1, g2, g3, g4, 0);
#endif
}

// A fragment (16x32 bf16): lane<16 -> M=lane, K = k0+{0..7,16..23};
// lane>=16 -> M=lane-16, K = k0+{8..15,24..31}
__device__ __forceinline__ v16bf frag_a(const u16* lds, int row, int k0, int koff) {
  const u16* p = lds + row * Hn + k0 + koff;
  union { uint4 q[2]; v16bf v; } u;
  u.q[0] = *(const uint4*)p;
  u.q[1] = *(const uint4*)(p + 16);
  return u.v;
}
// B fragment (32x16 bf16) from row-major W[n][k]: lane<16 -> N=lane, K = k0+{0..7,16..23};
// lane>=16 -> N=lane-16, K = k0+{8..15,24..31}
__device__ __forceinline__ v16bf frag_b(const u16* lds, int row, int k0, int hi) {
  const u16* p = lds + row * Hn + k0 + hi * 8;
  union { uint4 q[2]; v16bf v; } u;
  u.q[0] = *(const uint4*)p;
  u.q[1] = *(const uint4*)(p + 16);
  return u.v;
}

// ---------------- prep: fp32 weights -> bf16, zero per-column h-sums ----------------
__global__ void stlstm_prep(const float* __restrict__ W_th, const float* __restrict__ W_sh,
                            u16* __restrict__ wt_bf, u16* __restrict__ ws_bf,
                            float* __restrict__ hsum_col)
{
  const int NW = GH * Hn;       // 327680
  const int NS = Jn * Bn * Hn;  // 819200
  for (int i = blockIdx.x * blockDim.x + threadIdx.x; i < NS;
       i += gridDim.x * blockDim.x) {
    if (i < NW) {
      wt_bf[i] = f2bf(W_th[i]);
      ws_bf[i] = f2bf(W_sh[i]);
    }
    hsum_col[i] = 0.0f;
  }
}

// ---------------- one diagonal of the 2-D recurrence ----------------
// block (slice, j); t = d - j. Column state double-buffered by diagonal parity.
__global__ __launch_bounds__(256, 1)
void stlstm_step(const float* __restrict__ data, const float* __restrict__ W_ih,
                 const float* __restrict__ bias, const u16* __restrict__ wt_bf,
                 const u16* __restrict__ ws_bf, u16* __restrict__ h_col,
                 float* __restrict__ c_col, float* __restrict__ hsum_col, int d)
{
  const int j = blockIdx.y;
  const int t = d - j;
  if (t < 0 || t >= Tn) return;          // whole block exits -> EXEC full below

  const int slice = blockIdx.x;
  const int h0 = slice * HS;
  const int tid = (int)threadIdx.x;
  const int pr = (d + 1) & 1;            // parents were written on diagonal d-1
  const int pw = d & 1;

  extern __shared__ __align__(16) u16 smem[];
  u16* Ht = smem;                        // 128*256 bf16 (64 KB)
  u16* Hs = Ht + Bn * Hn;                // 64 KB
  u16* Wt = Hs + Bn * Hn;                // 160*256 bf16 (80 KB)
  u16* Ws = Wt + NCOL * Hn;              // 80 KB   -> total 288 KB

  // ---- stage this block's W_th / W_sh row slices via the Tensor Data Mover ----
#if __has_builtin(__builtin_amdgcn_tensor_load_to_lds)
  if (tid < 32) {                        // wave 0 issues both TDM descriptors
    tdm_load_w_slice(Wt, wt_bf + (size_t)h0 * Hn);
    tdm_load_w_slice(Ws, ws_bf + (size_t)h0 * Hn);
  }
#else
  {
#pragma unroll
    for (int i = 0; i < 20; ++i) {       // 5120 16B-chunks / matrix / 256 thr
      int c = i * 256 + tid;
      int r = c >> 5;                    // local gate-col 0..159
      int q = c & 31;                    // 16B chunk within the K row
      int n = (r >> 5) * Hn + h0 + (r & 31);
      cp_async16(Wt + r * Hn + q * 8, wt_bf + (size_t)n * Hn + q * 8);
      cp_async16(Ws + r * Hn + q * 8, ws_bf + (size_t)n * Hn + q * 8);
    }
  }
#endif

  // ---- stage temporal/spatial H (or zeros at boundaries), ASYNCcnt path ----
  {
    const uint4 z4 = make_uint4(0u, 0u, 0u, 0u);
    const u16* srcT = h_col + (size_t)(pr * Jn + j) * (Bn * Hn);
    const u16* srcS = h_col + (size_t)(pr * Jn + (j - 1)) * (Bn * Hn);
#pragma unroll
    for (int i = 0; i < 16; ++i) {       // 4096 16B-chunks / matrix / 256 thr
      int c = i * 256 + tid;
      if (t == 0) ((uint4*)Ht)[c] = z4;
      else        cp_async16(Ht + c * 8, srcT + c * 8);
      if (j == 0) ((uint4*)Hs)[c] = z4;
      else        cp_async16(Hs + c * 8, srcS + c * 8);
    }
  }
  async_wait_all();
#if __has_builtin(__builtin_amdgcn_tensor_load_to_lds)
  if (tid < 32) __builtin_amdgcn_s_wait_tensorcnt(0);
#endif
  __syncthreads();

  // ---- WMMA GEMM: each wave owns a 16-row band, 10 column tiles ----
  const int wave = tid >> 5;
  const int lane = tid & 31;
  const int r16  = lane & 15;
  const int hi   = lane >> 4;
  const int koff = hi * 8;
  const int m0   = wave * 16;

  v8f acc[NT];
  const float* xrow = data + (size_t)(t * Jn + j) * Bn * In;

#pragma unroll
  for (int nt = 0; nt < NT; ++nt) {
    // accumulator init: bias + x @ W_ih^T (K=3, scalar)
    const int g = nt >> 1;
    const int n = g * Hn + h0 + (nt & 1) * 16 + r16;
    const float w0 = W_ih[n * 3 + 0], w1 = W_ih[n * 3 + 1], w2 = W_ih[n * 3 + 2];
    const float bn = bias[n];
    v8f a;
#pragma unroll
    for (int v = 0; v < 8; ++v) {
      const int m = m0 + v + hi * 8;
      const float* x = xrow + m * 3;
      a[v] = bn + x[0] * w0 + x[1] * w1 + x[2] * w2;
    }
#pragma unroll
    for (int kk = 0; kk < 8; ++kk) {
      const int k0 = kk * 32;
      v16bf at = frag_a(Ht, m0 + r16, k0, koff);
      v16bf bt = frag_b(Wt, nt * 16 + r16, k0, hi);
      a = __builtin_amdgcn_wmma_f32_16x16x32_bf16(false, at, false, bt,
                                                  (short)0, a, false, false);
      v16bf as = frag_a(Hs, m0 + r16, k0, koff);
      v16bf bs = frag_b(Ws, nt * 16 + r16, k0, hi);
      a = __builtin_amdgcn_wmma_f32_16x16x32_bf16(false, as, false, bs,
                                                  (short)0, a, false, false);
    }
    acc[nt] = a;
  }

  // ---- elementwise ST-LSTM cell update, all in registers ----
  // lane holds gate values for hidden hh = p*16 + r16 at tiles {2g+p}
  const size_t cbT = (size_t)(pr * Jn + j) * Bn * Hn;
  const size_t cbS = (size_t)(pr * Jn + (j - 1)) * Bn * Hn;
  const size_t bW  = (size_t)(pw * Jn + j) * Bn * Hn;
  const size_t sb  = (size_t)j * Bn * Hn;

#pragma unroll
  for (int p = 0; p < 2; ++p) {
    const int hidx = h0 + p * 16 + r16;
#pragma unroll
    for (int v = 0; v < 8; ++v) {
      const int m = m0 + v + hi * 8;
      const size_t o = (size_t)m * Hn + hidx;
      const float ct = (t == 0) ? 0.0f : c_col[cbT + o];
      const float cs = (j == 0) ? 0.0f : c_col[cbS + o];
      const float gi = acc[0 + p][v], gfs = acc[2 + p][v], gft = acc[4 + p][v],
                  go = acc[6 + p][v], gu  = acc[8 + p][v];
      const float c = sigmoidf_(gi) * tanhf(gu)
                    + sigmoidf_(gfs) * cs + sigmoidf_(gft) * ct;
      const float h = sigmoidf_(go) * tanhf(c);
      c_col[bW + o]  = c;
      h_col[bW + o]  = f2bf(h);
      hsum_col[sb + o] += h;     // unique (j,m,hidx) per block -> no race
    }
  }
}

// ---------------- classifier head: mean -> logits -> log_softmax ----------------
__global__ __launch_bounds__(64)
void stlstm_head(const float* __restrict__ hsum_col, const float* __restrict__ weight_c,
                 const float* __restrict__ bias_c, float* __restrict__ out)
{
  const int b = blockIdx.x;
  const int tid = (int)threadIdx.x;
  __shared__ float F[Hn];
  __shared__ float logits[Cn];
  __shared__ float red[2];

  for (int h = tid; h < Hn; h += 64) {
    float s = 0.0f;
    for (int j = 0; j < Jn; ++j) s += hsum_col[((size_t)j * Bn + b) * Hn + h];
    F[h] = s * (1.0f / (float)(Tn * Jn));
  }
  __syncthreads();
  for (int c = tid; c < Cn; c += 64) {
    float a = bias_c[c];
    for (int h = 0; h < Hn; ++h) a += F[h] * weight_c[(size_t)c * Hn + h];
    logits[c] = a;
  }
  __syncthreads();
  if (tid == 0) {
    float mx = -1e30f;
    for (int c = 0; c < Cn; ++c) mx = fmaxf(mx, logits[c]);
    float s = 0.0f;
    for (int c = 0; c < Cn; ++c) s += __expf(logits[c] - mx);
    red[0] = mx;
    red[1] = __logf(s);
  }
  __syncthreads();
  for (int c = tid; c < Cn; c += 64)
    out[(size_t)b * Cn + c] = logits[c] - red[0] - red[1];
}

extern "C" void kernel_launch(void* const* d_in, const int* in_sizes, int n_in,
                              void* d_out, int out_size, void* d_ws, size_t ws_size,
                              hipStream_t stream)
{
  const float* data = (const float*)d_in[0];   // (T,J,B,I)
  const float* W_ih = (const float*)d_in[1];   // (5H,I)
  const float* W_th = (const float*)d_in[2];   // (5H,H)
  const float* W_sh = (const float*)d_in[3];   // (5H,H)
  const float* bias = (const float*)d_in[4];   // (5H,)
  const float* w_c  = (const float*)d_in[5];   // (C,H)
  const float* b_c  = (const float*)d_in[6];   // (C,)
  float* out = (float*)d_out;

  // workspace layout (bytes): total ~13.8 MB
  char* ws = (char*)d_ws;
  u16*   wt_bf = (u16*)(ws + 0);                 //  655360
  u16*   ws_bf = (u16*)(ws + 655360);            //  655360
  u16*   h_col = (u16*)(ws + 1310720);           // 3276800 (2 parities x J x B x H bf16)
  float* c_col = (float*)(ws + 4587520);         // 6553600 (2 parities x J x B x H f32)
  float* hsum  = (float*)(ws + 11141120);        // 3276800 (J x B x H f32)

  const size_t smem = (size_t)(2 * Bn * Hn + 2 * NCOL * Hn) * sizeof(u16); // 294912 B
  (void)hipFuncSetAttribute((const void*)stlstm_step,
                            hipFuncAttributeMaxDynamicSharedMemorySize, (int)smem);

  stlstm_prep<<<400, 256, 0, stream>>>(W_th, W_sh, wt_bf, ws_bf, hsum);

  for (int dd = 0; dd < Tn + Jn - 1; ++dd) {     // 124 diagonals
    stlstm_step<<<dim3(NSLICE, Jn), 256, smem, stream>>>(
        data, W_ih, bias, wt_bf, ws_bf, h_col, c_col, hsum, dd);
  }

  stlstm_head<<<Bn, 64, 0, stream>>>(hsum, w_c, b_c, out);
}
